// SpatioTemporalTransformer_68255620268645
// MI455X (gfx1250) — compile-verified, matrix operand order checked by reading, not run on hardware
//
#include <hip/hip_runtime.h>

// ---------------- constants ----------------
#define Bsz   16
#define Sseq  256
#define Dmod  512
#define Hh    8
#define DH    64
#define FFd   2048
#define Ll    4
#define NPOI  5000
#define NPOI_PAD 5024
#define HIDm  8
#define Mrows (Bsz*Sseq)   // 4096
#define TPITCH 40          // LDS row pitch in bf16 elems (80B = 20 banks, conflict-free frag reads)

typedef __attribute__((ext_vector_type(16))) __bf16 v16bf;
typedef __attribute__((ext_vector_type(8)))  __bf16 v8bf;
typedef __attribute__((ext_vector_type(8)))  float  v8f;

// ---------------- helpers ----------------
__device__ __forceinline__ __bf16 f2bf(float f) {
    union { float f; unsigned u; } a; a.f = f;
    unsigned r = a.u + 0x7fffu + ((a.u >> 16) & 1u);   // round-to-nearest-even
    unsigned short h = (unsigned short)(r >> 16);
    __bf16 out;
    __builtin_memcpy(&out, &h, 2);
    return out;
}

// Load one 16x32 bf16 WMMA fragment from a row-major tile (ld = pitch elems).
// Lane l<16: row=l, K-chunks [kb..kb+7],[kb+16..kb+23]; lanes>=16: kb += 8.
__device__ __forceinline__ v16bf load_frag16(const __bf16* p) {
    v8bf lo = *reinterpret_cast<const v8bf*>(p);
    v8bf hi = *reinterpret_cast<const v8bf*>(p + 16);
    v16bf f;
#pragma unroll
    for (int i = 0; i < 8; ++i) { f[i] = lo[i]; f[i + 8] = hi[i]; }
    return f;
}

// ---------------- WMMA GEMM (LDS double-buffered) ----------------
// C[M,N] (f32, ld=N) = alpha * A[M,K] x Bt[N,K]^T + bias[N], optional relu.
// A is f32 (AF32=true, converted to bf16 while staging) or bf16.
// Block: 256 thr = 8 waves; block tile 128x128; wave tile 32x64 (2x4 WMMA 16x16x32).
// Double-buffered LDS tiles (A:128x32, B:128x32 bf16, pitch 40); one barrier per K-chunk.
template<bool AF32>
__global__ __launch_bounds__(256)
void gemm_wmma_lds(const void* __restrict__ Ap, const __bf16* __restrict__ BtBase,
                   float* __restrict__ C, const float* __restrict__ bias,
                   int M, int N, int K, int NbRows,
                   long long sA, long long sB, long long sC,
                   float alpha, int relu)
{
    __shared__ __bf16 Abuf[2][128 * TPITCH];
    __shared__ __bf16 Bbuf[2][128 * TPITCH];

    const int bat = blockIdx.z;
    const float*  Af = (const float*)Ap  + (AF32 ? (long long)bat * sA : 0);
    const __bf16* Ab = (const __bf16*)Ap + (AF32 ? 0 : (long long)bat * sA);
    const __bf16* Bt = BtBase + (long long)bat * sB;
    C += (long long)bat * sC;

    const int t     = threadIdx.x;
    const int lane  = t & 31;
    const int wid   = t >> 5;
    const int wr    = wid & 3;          // 0..3 : 32-row slice
    const int wc    = wid >> 2;         // 0..1 : 64-col slice
    const int mBase = blockIdx.y * 128;
    const int nBase = blockIdx.x * 128;
    const int r     = lane & 15;
    const int halfk = (lane >> 4) * 8;

    v8f acc[2][4];
#pragma unroll
    for (int i = 0; i < 2; ++i)
#pragma unroll
        for (int j = 0; j < 4; ++j) acc[i][j] = (v8f){};

    // stage one 128x32 A tile + 128x32 B tile into LDS buffer `buf` at k-offset k0.
    // 512 chunks of 8 elems each per tile; 2 chunks per thread per tile.
    auto stageTo = [&](int buf, int k0) {
#pragma unroll
        for (int c = 0; c < 2; ++c) {
            const int ch = t * 2 + c, row = ch >> 2, co = (ch & 3) * 8;
            v8bf o;
            if (AF32) {
                const float4* p = (const float4*)(Af + (long long)(mBase + row) * K + k0 + co);
                float4 fa = p[0], fb = p[1];
                o[0] = f2bf(fa.x); o[1] = f2bf(fa.y); o[2] = f2bf(fa.z); o[3] = f2bf(fa.w);
                o[4] = f2bf(fb.x); o[5] = f2bf(fb.y); o[6] = f2bf(fb.z); o[7] = f2bf(fb.w);
            } else {
                o = *(const v8bf*)(Ab + (long long)(mBase + row) * K + k0 + co);
            }
            *(v8bf*)&Abuf[buf][row * TPITCH + co] = o;
        }
#pragma unroll
        for (int c = 0; c < 2; ++c) {
            const int ch = t * 2 + c, row = ch >> 2, co = (ch & 3) * 8;
            const int gn = nBase + row;
            v8bf o = {};
            if (gn < NbRows) o = *(const v8bf*)(Bt + (long long)gn * K + k0 + co);
            *(v8bf*)&Bbuf[buf][row * TPITCH + co] = o;
        }
    };

    const int nk = K >> 5;
    stageTo(0, 0);
    __syncthreads();

    for (int kc = 0; kc < nk; ++kc) {
        const int cur = kc & 1;
        if (kc + 1 < nk) stageTo(cur ^ 1, (kc + 1) << 5);   // overlap next-tile global loads with WMMA

        v16bf afrag[2], bfrag[4];
#pragma unroll
        for (int i = 0; i < 2; ++i)
            afrag[i] = load_frag16(&Abuf[cur][(wr * 32 + 16 * i + r) * TPITCH + halfk]);
#pragma unroll
        for (int j = 0; j < 4; ++j)
            bfrag[j] = load_frag16(&Bbuf[cur][(wc * 64 + 16 * j + r) * TPITCH + halfk]);
#pragma unroll
        for (int i = 0; i < 2; ++i)
#pragma unroll
            for (int j = 0; j < 4; ++j)
                acc[i][j] = __builtin_amdgcn_wmma_f32_16x16x32_bf16(
                    false, afrag[i], false, bfrag[j], (short)0, acc[i][j], false, false);

        __syncthreads();
    }

    const int rofs = (lane >> 4) * 8;   // C layout: lanes 0-15 -> M=v, lanes 16-31 -> M=8+v
#pragma unroll
    for (int ti = 0; ti < 2; ++ti) {
#pragma unroll
        for (int tj = 0; tj < 4; ++tj) {
            const int col = nBase + wc * 64 + tj * 16 + r;
            if (col < N) {
                const float bv = bias ? bias[col] : 0.0f;
#pragma unroll
                for (int v = 0; v < 8; ++v) {
                    const int row = mBase + wr * 32 + ti * 16 + rofs + v;
                    float c = acc[ti][tj][v] * alpha + bv;
                    if (relu) c = fmaxf(c, 0.0f);
                    C[(long long)row * N + col] = c;
                }
            }
        }
    }
}

// ---------------- small kernels ----------------
__global__ __launch_bounds__(256)
void add_pe_kernel(const float* __restrict__ x, const float* __restrict__ pe,
                   float* __restrict__ xw)
{
    long long i = (long long)blockIdx.x * 256 + threadIdx.x;
    if (i >= (long long)Mrows * Dmod) return;
    int d = (int)(i % Dmod);
    int b = (int)(i / ((long long)Sseq * Dmod));
    xw[i] = x[i] + pe[(long long)b * Dmod + d];   // faithful quirk: pe indexed by batch
}

// w (K,N) f32  ->  wt (Npad,K) bf16  (B transposed; zero pad rows n>=N)
__global__ __launch_bounds__(256)
void weight_t_bf16_kernel(const float* __restrict__ w, __bf16* __restrict__ wt,
                          int K, int N, int Npad)
{
    long long i = (long long)blockIdx.x * 256 + threadIdx.x;
    if (i >= (long long)Npad * K) return;
    int n = (int)(i / K);
    int k = (int)(i % K);
    wt[i] = (n < N) ? f2bf(w[(long long)k * N + n]) : f2bf(0.0f);
}

// (B,S,D) f32 q/k/v -> q,k bf16 (B,H,S,DH); v bf16 transposed (B,H,DH,S)
__global__ __launch_bounds__(256)
void pack_qkv_kernel(const float* __restrict__ tq, const float* __restrict__ tk,
                     const float* __restrict__ tv,
                     __bf16* __restrict__ q, __bf16* __restrict__ k, __bf16* __restrict__ vt)
{
    long long i = (long long)blockIdx.x * 256 + threadIdx.x;
    if (i >= (long long)Mrows * Dmod) return;
    int d  = (int)(i % Dmod);
    long long bs = i / Dmod;
    int s = (int)(bs % Sseq);
    int b = (int)(bs / Sseq);
    int h = d >> 6, dh = d & 63;
    long long qi = (((long long)(b * Hh + h)) * Sseq + s) * DH + dh;
    q[qi] = f2bf(tq[i]);
    k[qi] = f2bf(tk[i]);
    long long vi = (((long long)(b * Hh + h)) * DH + dh) * Sseq + s;
    vt[vi] = f2bf(tv[i]);
}

// scores(b,h,i,j) += MLP_t(td[b,i,j]) + MLP_d(dd[b,i,j]); 1->8->8 MLPs
__global__ __launch_bounds__(256)
void bias_mlp_add_kernel(float* __restrict__ scores,
                         const float* __restrict__ td, const float* __restrict__ dd,
                         const float* __restrict__ tw1, const float* __restrict__ tb1,
                         const float* __restrict__ tw2, const float* __restrict__ tb2,
                         const float* __restrict__ dw1, const float* __restrict__ db1,
                         const float* __restrict__ dw2, const float* __restrict__ db2)
{
    long long idx = (long long)blockIdx.x * 256 + threadIdx.x;
    if (idx >= (long long)Bsz * Sseq * Sseq) return;
    int j = (int)(idx & 255);
    int i = (int)((idx >> 8) & 255);
    int b = (int)(idx >> 16);
    float tv = td[idx], dv = dd[idx];
    float acc[Hh];
#pragma unroll
    for (int h = 0; h < Hh; ++h) acc[h] = tb2[h] + db2[h];
#pragma unroll
    for (int k = 0; k < HIDm; ++k) {
        float ht = fmaxf(tv * tw1[k] + tb1[k], 0.0f);
        float hd = fmaxf(dv * dw1[k] + db1[k], 0.0f);
#pragma unroll
        for (int h = 0; h < Hh; ++h) acc[h] += ht * tw2[k * Hh + h] + hd * dw2[k * Hh + h];
    }
#pragma unroll
    for (int h = 0; h < Hh; ++h) {
        long long si = ((((long long)b * Hh + h) * Sseq + i) << 8) + j;
        scores[si] += acc[h];
    }
}

// softmax over each row of length 256; output bf16
__global__ __launch_bounds__(256)
void softmax_rows_kernel(const float* __restrict__ scores, __bf16* __restrict__ attn)
{
    __shared__ float red[256];
    long long row = blockIdx.x;
    int t = threadIdx.x;
    float v = scores[row * 256 + t];
    red[t] = v; __syncthreads();
    for (int s = 128; s > 0; s >>= 1) { if (t < s) red[t] = fmaxf(red[t], red[t + s]); __syncthreads(); }
    float mx = red[0]; __syncthreads();
    float e = __expf(v - mx);
    red[t] = e; __syncthreads();
    for (int s = 128; s > 0; s >>= 1) { if (t < s) red[t] += red[t + s]; __syncthreads(); }
    attn[row * 256 + t] = f2bf(e / red[0]);
}

// x[row,:] = LN(x + add) * g + b    (in place; mode 0: no add, 1: plain (B,S,D), 2: headed (B,H,S,DH))
__global__ __launch_bounds__(256)
void resid_ln_kernel(float* __restrict__ x, const float* __restrict__ add, int mode,
                     const float* __restrict__ g, const float* __restrict__ bta)
{
    __shared__ float s1[256], s2[256];
    int row = blockIdx.x;        // b*S + s
    int t   = threadIdx.x;
    int b   = row >> 8;
    int s   = row & 255;
    float v[2];
#pragma unroll
    for (int e = 0; e < 2; ++e) {
        int d = t + e * 256;
        float xv = x[(long long)row * Dmod + d];
        if (mode == 1) xv += add[(long long)row * Dmod + d];
        else if (mode == 2) {
            int h = d >> 6, dh = d & 63;
            xv += add[((((long long)b * Hh + h) * Sseq + s) << 6) + dh];
        }
        v[e] = xv;
    }
    s1[t] = v[0] + v[1];
    s2[t] = v[0] * v[0] + v[1] * v[1];
    __syncthreads();
    for (int k = 128; k > 0; k >>= 1) { if (t < k) { s1[t] += s1[t + k]; s2[t] += s2[t + k]; } __syncthreads(); }
    float mean = s1[0] * (1.0f / Dmod);
    float var  = s2[0] * (1.0f / Dmod) - mean * mean;
    float inv  = rsqrtf(var + 1e-5f);
#pragma unroll
    for (int e = 0; e < 2; ++e) {
        int d = t + e * 256;
        x[(long long)row * Dmod + d] = (v[e] - mean) * inv * g[d] + bta[d];
    }
}

// ---------------- host orchestration ----------------
static inline unsigned g1d(long long n) { return (unsigned)((n + 255) / 256); }

extern "C" void kernel_launch(void* const* d_in, const int* in_sizes, int n_in,
                              void* d_out, int out_size, void* d_ws, size_t ws_size,
                              hipStream_t stream)
{
    (void)in_sizes; (void)n_in; (void)out_size; (void)ws_size;
    const float* x    = (const float*)d_in[0];
    const float* tdel = (const float*)d_in[1];
    const float* dst  = (const float*)d_in[2];
    const float* pe   = (const float*)d_in[3];
    const float* qw   = (const float*)d_in[4];  const float* qbias = (const float*)d_in[5];
    const float* kw   = (const float*)d_in[6];  const float* kbias = (const float*)d_in[7];
    const float* vw   = (const float*)d_in[8];  const float* vbias = (const float*)d_in[9];
    const float* tw1  = (const float*)d_in[10]; const float* tb1   = (const float*)d_in[11];
    const float* tw2  = (const float*)d_in[12]; const float* tb2   = (const float*)d_in[13];
    const float* dw1  = (const float*)d_in[14]; const float* db1   = (const float*)d_in[15];
    const float* dw2  = (const float*)d_in[16]; const float* db2   = (const float*)d_in[17];
    const float* fw1  = (const float*)d_in[18]; const float* fb1   = (const float*)d_in[19];
    const float* fw2  = (const float*)d_in[20]; const float* fb2   = (const float*)d_in[21];
    const float* g1   = (const float*)d_in[22]; const float* b1    = (const float*)d_in[23];
    const float* g2   = (const float*)d_in[24]; const float* b2    = (const float*)d_in[25];
    const float* gf   = (const float*)d_in[26]; const float* bf    = (const float*)d_in[27];
    const float* ow   = (const float*)d_in[28]; const float* ob    = (const float*)d_in[29];
    float* out = (float*)d_out;

    // workspace carve-up (256B aligned)
    char* ws = (char*)d_ws; size_t off = 0;
    auto carve = [&](size_t bytes) -> void* {
        void* p = ws + off;
        off += (bytes + 255) & ~(size_t)255;
        return p;
    };
    float*  xw     = (float*) carve((size_t)Mrows * Dmod * 4);          // working activations
    float*  tQ     = (float*) carve((size_t)Mrows * Dmod * 4);          // also attn-out / ffn2-out
    float*  tK     = (float*) carve((size_t)Mrows * Dmod * 4);
    float*  tV     = (float*) carve((size_t)Mrows * Dmod * 4);
    __bf16* qbf    = (__bf16*)carve((size_t)Mrows * Dmod * 2);
    __bf16* kbf    = (__bf16*)carve((size_t)Mrows * Dmod * 2);
    __bf16* vtbf   = (__bf16*)carve((size_t)Mrows * Dmod * 2);
    float*  scores = (float*) carve((size_t)Bsz * Hh * Sseq * Sseq * 4); // == M*FF*4 -> reused as FFN hidden f32
    __bf16* attnb  = (__bf16*)carve((size_t)Bsz * Hh * Sseq * Sseq * 2);
    __bf16* qwt    = (__bf16*)carve((size_t)Ll * Dmod * Dmod * 2);
    __bf16* kwt    = (__bf16*)carve((size_t)Ll * Dmod * Dmod * 2);
    __bf16* vwt    = (__bf16*)carve((size_t)Ll * Dmod * Dmod * 2);
    __bf16* fw1t   = (__bf16*)carve((size_t)Ll * FFd * Dmod * 2);
    __bf16* fw2t   = (__bf16*)carve((size_t)Ll * Dmod * FFd * 2);
    __bf16* owt    = (__bf16*)carve((size_t)NPOI_PAD * Dmod * 2);
    float*  hidf   = scores;

    auto gemm = [&](const void* A, bool af32, const __bf16* Bt, float* C, const float* bias,
                    int M, int N, int K, int NbRows,
                    long long sA, long long sB, long long sC, int batch, float alpha, int relu) {
        dim3 grid((N + 127) / 128, M / 128, batch);
        if (af32)
            gemm_wmma_lds<true ><<<grid, 256, 0, stream>>>(A, Bt, C, bias, M, N, K, NbRows, sA, sB, sC, alpha, relu);
        else
            gemm_wmma_lds<false><<<grid, 256, 0, stream>>>(A, Bt, C, bias, M, N, K, NbRows, sA, sB, sC, alpha, relu);
    };

    // --- convert / transpose all weights to bf16 (N,K) ---
    for (int l = 0; l < Ll; ++l) {
        weight_t_bf16_kernel<<<g1d((long long)Dmod * Dmod), 256, 0, stream>>>(qw + (size_t)l*Dmod*Dmod, qwt + (size_t)l*Dmod*Dmod, Dmod, Dmod, Dmod);
        weight_t_bf16_kernel<<<g1d((long long)Dmod * Dmod), 256, 0, stream>>>(kw + (size_t)l*Dmod*Dmod, kwt + (size_t)l*Dmod*Dmod, Dmod, Dmod, Dmod);
        weight_t_bf16_kernel<<<g1d((long long)Dmod * Dmod), 256, 0, stream>>>(vw + (size_t)l*Dmod*Dmod, vwt + (size_t)l*Dmod*Dmod, Dmod, Dmod, Dmod);
        weight_t_bf16_kernel<<<g1d((long long)FFd * Dmod), 256, 0, stream>>>(fw1 + (size_t)l*Dmod*FFd, fw1t + (size_t)l*FFd*Dmod, Dmod, FFd, FFd);
        weight_t_bf16_kernel<<<g1d((long long)Dmod * FFd), 256, 0, stream>>>(fw2 + (size_t)l*FFd*Dmod, fw2t + (size_t)l*Dmod*FFd, FFd, Dmod, Dmod);
    }
    weight_t_bf16_kernel<<<g1d((long long)NPOI_PAD * Dmod), 256, 0, stream>>>(ow, owt, Dmod, NPOI, NPOI_PAD);

    // --- x = x + pe[:B] ---
    add_pe_kernel<<<g1d((long long)Mrows * Dmod), 256, 0, stream>>>(x, pe, xw);

    const float scale = 1.0f / 22.627416997969522f;   // 1/sqrt(512)

    for (int l = 0; l < Ll; ++l) {
        // QKV projections (A = xw f32, converted in-GEMM)
        gemm(xw, true, qwt + (size_t)l*Dmod*Dmod, tQ, qbias + (size_t)l*Dmod, Mrows, Dmod, Dmod, Dmod, 0, 0, 0, 1, 1.0f, 0);
        gemm(xw, true, kwt + (size_t)l*Dmod*Dmod, tK, kbias + (size_t)l*Dmod, Mrows, Dmod, Dmod, Dmod, 0, 0, 0, 1, 1.0f, 0);
        gemm(xw, true, vwt + (size_t)l*Dmod*Dmod, tV, vbias + (size_t)l*Dmod, Mrows, Dmod, Dmod, Dmod, 0, 0, 0, 1, 1.0f, 0);
        pack_qkv_kernel<<<g1d((long long)Mrows * Dmod), 256, 0, stream>>>(tQ, tK, tV, qbf, kbf, vtbf);

        // scores = scale * q @ k^T  (batched over B*H)
        gemm(qbf, false, kbf, scores, nullptr, Sseq, Sseq, DH, Sseq,
             (long long)Sseq*DH, (long long)Sseq*DH, (long long)Sseq*Sseq,
             Bsz*Hh, scale, 0);

        // += temporal + distance bias MLPs
        bias_mlp_add_kernel<<<g1d((long long)Bsz*Sseq*Sseq), 256, 0, stream>>>(
            scores, tdel, dst,
            tw1 + (size_t)l*HIDm, tb1 + (size_t)l*HIDm, tw2 + (size_t)l*HIDm*Hh, tb2 + (size_t)l*Hh,
            dw1 + (size_t)l*HIDm, db1 + (size_t)l*HIDm, dw2 + (size_t)l*HIDm*Hh, db2 + (size_t)l*Hh);

        // softmax -> attn bf16
        softmax_rows_kernel<<<Bsz*Hh*Sseq, 256, 0, stream>>>(scores, attnb);

        // out = attn @ v   (batched; vt (B,H,DH,S) == B^T per head; only 64 valid B rows)
        gemm(attnb, false, vtbf, tQ, nullptr, Sseq, DH, Sseq, DH,
             (long long)Sseq*Sseq, (long long)DH*Sseq, (long long)Sseq*DH,
             Bsz*Hh, 1.0f, 0);

        // x = LN(x + attn_out)
        resid_ln_kernel<<<Mrows, 256, 0, stream>>>(xw, tQ, 2, g1 + (size_t)l*Dmod, b1 + (size_t)l*Dmod);

        // FFN (A f32 both times; hidden stays f32, converted in-GEMM)
        gemm(xw, true, fw1t + (size_t)l*FFd*Dmod, hidf, fb1 + (size_t)l*FFd, Mrows, FFd, Dmod, FFd, 0, 0, 0, 1, 1.0f, 1);
        gemm(hidf, true, fw2t + (size_t)l*Dmod*FFd, tQ, fb2 + (size_t)l*Dmod, Mrows, Dmod, FFd, Dmod, 0, 0, 0, 1, 1.0f, 0);

        // x = LN(x + ffn)
        resid_ln_kernel<<<Mrows, 256, 0, stream>>>(xw, tQ, 1, g2 + (size_t)l*Dmod, b2 + (size_t)l*Dmod);
    }

    // final LN, then projection to NPOI
    resid_ln_kernel<<<Mrows, 256, 0, stream>>>(xw, nullptr, 0, gf, bf);
    gemm(xw, true, owt, out, ob, Mrows, NPOI, Dmod, NPOI_PAD, 0, 0, 0, 1, 1.0f, 0);
}
